// CrossAttention_90735479095348
// MI455X (gfx1250) — matrix operand order repaired and checked by reading, not compiled, not run on hardware
//
#include <hip/hip_runtime.h>
#include <hip/hip_bf16.h>

// ---------------------------------------------------------------------------
// Cross-attention, fused, bf16 WMMA (gfx1250 / CDNA5, wave32)
// B=8 T=4096 S=77 hidden=embed=640 ctx=768 H=8 DH=80
// 32-row M-tiles per WG: each weight B-fragment feeds 2 WMMAs (halves L2 traffic)
// ---------------------------------------------------------------------------

typedef __attribute__((ext_vector_type(16))) __bf16        v16bf;
typedef __attribute__((ext_vector_type(8)))  float         v8f;
typedef __attribute__((ext_vector_type(4)))  unsigned int  v4u;
typedef __attribute__((ext_vector_type(4)))  float         v4f;

#define HID 640
#define CTXD 768
#define SEQ 77
#define NH 8
#define DH 80
#define T_TOT 4096
#define B_TOT 8
#define MROWS 32               // rows per workgroup

union FragB16 {
    v16bf bf;
    v4u   u[2];
    unsigned short us[16];
};

__device__ __forceinline__ unsigned short f2bf(float x) {
    unsigned int u = __builtin_bit_cast(unsigned int, x);
    u += 0x7FFFu + ((u >> 16) & 1u);            // round-to-nearest-even
    return (unsigned short)(u >> 16);
}

__device__ __forceinline__ unsigned int pack2bf(float lo, float hi) {
    return ((unsigned int)f2bf(hi) << 16) | (unsigned int)f2bf(lo);
}

__device__ __forceinline__ v8f wmma_bf16(const FragB16& A, const FragB16& B, v8f C) {
    return __builtin_amdgcn_wmma_f32_16x16x32_bf16(false, A.bf, false, B.bf,
                                                   (short)0, C, false, false);
}

// ---------------------------------------------------------------------------
// Kernel 1: convert + transpose weights to bf16.
//   WqT[n][k] = Wq[k][n]  (640x640)   WoT[n][k] = Wo[k][n]  (640x640)
//   WkT[n][k] = Wk[k][n]  (640x768)   WvT[n][k] = Wv[k][n]  (640x768)
// ---------------------------------------------------------------------------
__global__ __launch_bounds__(256) void ca_prep_weights(
    const float* __restrict__ Wq, const float* __restrict__ Wk,
    const float* __restrict__ Wv, const float* __restrict__ Wo,
    unsigned short* __restrict__ WqT, unsigned short* __restrict__ WkT,
    unsigned short* __restrict__ WvT, unsigned short* __restrict__ WoT) {
    int idx = blockIdx.x * 256 + threadIdx.x;
    if (idx >= HID * CTXD) return;
    int n = idx / CTXD;
    int k = idx % CTXD;
    WkT[n * CTXD + k] = f2bf(Wk[k * HID + n]);
    WvT[n * CTXD + k] = f2bf(Wv[k * HID + n]);
    if (k < HID) {
        WqT[n * HID + k] = f2bf(Wq[k * HID + n]);
        WoT[n * HID + k] = f2bf(Wo[k * HID + n]);
    }
}

// ---------------------------------------------------------------------------
// Kernel 2: K/V projections via WMMA.
//   Kc[b][h][s(80)][d(96 pad)]  : B-operand for scores (lane = s, contiguous d)
//   Vc[b][h][d(80)][s(96 pad)]  : B-operand for ctx    (lane = d, contiguous s)
// Padding regions pre-zeroed by hipMemsetAsync.
// ---------------------------------------------------------------------------
__global__ __launch_bounds__(256) void ca_kv_kernel(
    const float* __restrict__ context,
    const unsigned short* __restrict__ WkT, const unsigned short* __restrict__ WvT,
    unsigned short* __restrict__ Kc, unsigned short* __restrict__ Vc) {
    int b = blockIdx.x >> 3;
    int h = blockIdx.x & 7;
    int wave = threadIdx.x >> 5;
    int lane = threadIdx.x & 31;
    int hi = lane >> 4, lq = lane & 15;
    int job = blockIdx.y * 8 + wave;        // 0..49 (25 K-tiles, 25 V-tiles)
    if (job >= 50) return;
    int mat = job / 25;                     // 0 = K, 1 = V
    int tile = job % 25;
    int st = tile / 5;                      // s-tile (M)
    int dt = tile % 5;                      // d-tile (N)
    const float* cb = context + (size_t)b * SEQ * CTXD;
    const unsigned short* WT = mat ? WvT : WkT;
    const unsigned short* brow = WT + (size_t)(h * DH + dt * 16 + lq) * CTXD;
    int s_a = st * 16 + lq;                 // A-row = sequence index for this lane

    v8f acc = (v8f)0.0f;
    for (int ks = 0; ks < CTXD / 32; ++ks) {
        int kb = ks * 32;
        FragB16 A, B;
        alignas(16) float av[16];
        if (s_a < SEQ) {
            const float* row = cb + (size_t)s_a * CTXD + kb + 8 * hi;
            *(v4f*)&av[0]  = *(const v4f*)(row);
            *(v4f*)&av[4]  = *(const v4f*)(row + 4);
            *(v4f*)&av[8]  = *(const v4f*)(row + 16);
            *(v4f*)&av[12] = *(const v4f*)(row + 20);
        } else {
#pragma unroll
            for (int e = 0; e < 16; ++e) av[e] = 0.0f;
        }
#pragma unroll
        for (int e = 0; e < 16; ++e) A.us[e] = f2bf(av[e]);
        const unsigned short* bp = brow + kb + 8 * hi;
        B.u[0] = *(const v4u*)(bp);
        B.u[1] = *(const v4u*)(bp + 16);
        acc = wmma_bf16(A, B, acc);
    }

    unsigned short* base = (mat ? Vc : Kc) + (size_t)(b * NH + h) * DH * 96;
#pragma unroll
    for (int v = 0; v < 8; ++v) {
        int s = st * 16 + v + 8 * hi;       // output row (sequence)
        int d = dt * 16 + lq;               // output col (head dim)
        unsigned short val = f2bf(acc[v]);
        if (mat == 0) {
            base[s * 96 + d] = val;         // Kc[s][d], s < 80 always
        } else if (s < SEQ) {
            base[d * 96 + s] = val;         // Vc[d][s], keep s>=77 zero
        }
    }
}

// ---------------------------------------------------------------------------
// Kernel 3: fused per-32-row tile:
//   Q-proj (both 16-row subtiles, shared B) -> per-subtile {scores, softmax,
//   ctx} -> out-proj (both subtiles, shared B).
// 8 waves; wave h owns head h for Q/attn, and output columns [80h, 80h+80).
// LDS: tok (32x640 bf16, tokens then ctx) + qb (16x768, single-wave stripes).
// ---------------------------------------------------------------------------
__global__ __launch_bounds__(256) void ca_attn_kernel(
    const float* __restrict__ tokens, const float* __restrict__ bo,
    const unsigned short* __restrict__ WqT, const unsigned short* __restrict__ WoT,
    const unsigned short* __restrict__ Kc, const unsigned short* __restrict__ Vc,
    float* __restrict__ out) {
    __shared__ alignas(16) unsigned short tok[MROWS * HID];  // 40 KB: tokens, later ctx
    __shared__ alignas(16) unsigned short qb[16 * 768];      // 24 KB: per-head stripes

    int b  = blockIdx.x >> 7;                    // 128 tiles of 32 rows per batch
    int t0 = (blockIdx.x & 127) * MROWS;
    int tid = threadIdx.x;
    int h = tid >> 5;
    int lane = tid & 31;
    int hi = lane >> 4, lq = lane & 15;

    // ---- stage 32x640 token tile as bf16 into LDS (packed, coalesced) ----
    const float* tb = tokens + ((size_t)(b * T_TOT + t0)) * HID;
    {
        unsigned int* tok32 = (unsigned int*)tok;
        for (int i4 = tid; i4 < MROWS * HID / 4; i4 += 256) {
            v4f f = *(const v4f*)(tb + (size_t)i4 * 4);
            tok32[i4 * 2]     = pack2bf(f.x, f.y);
            tok32[i4 * 2 + 1] = pack2bf(f.z, f.w);
        }
    }
    __syncthreads();

    // ---- Q projection: wave h computes Q[:, 80h : 80h+80] for both subtiles ----
    v8f qacc[2][5];
#pragma unroll
    for (int s = 0; s < 2; ++s)
#pragma unroll
        for (int n = 0; n < 5; ++n) qacc[s][n] = (v8f)0.0f;

    const unsigned short* bq[5];
#pragma unroll
    for (int nt = 0; nt < 5; ++nt)
        bq[nt] = WqT + (size_t)(h * DH + nt * 16 + lq) * HID + 8 * hi;

    for (int ks = 0; ks < HID / 32; ++ks) {
        int kb = ks * 32;
        FragB16 A0, A1;
        const unsigned short* ap0 = &tok[lq * HID + kb + 8 * hi];
        const unsigned short* ap1 = &tok[(16 + lq) * HID + kb + 8 * hi];
        A0.u[0] = *(const v4u*)(ap0);
        A0.u[1] = *(const v4u*)(ap0 + 16);
        A1.u[0] = *(const v4u*)(ap1);
        A1.u[1] = *(const v4u*)(ap1 + 16);
#pragma unroll
        for (int nt = 0; nt < 5; ++nt) {
            FragB16 B;
            B.u[0] = *(const v4u*)(bq[nt] + kb);
            B.u[1] = *(const v4u*)(bq[nt] + kb + 16);
            qacc[0][nt] = wmma_bf16(A0, B, qacc[0][nt]);
            qacc[1][nt] = wmma_bf16(A1, B, qacc[1][nt]);
        }
    }
    // All waves must finish reading the token tile before any wave's ctx
    // subtile overwrites it below.
    __syncthreads();

    const float scale = 0.11180339887498949f;    // 1/sqrt(80)
    const unsigned short* Kb = Kc + (size_t)(b * NH + h) * DH * 96;
    const unsigned short* Vb = Vc + (size_t)(b * NH + h) * DH * 96;

#pragma unroll
    for (int sub = 0; sub < 2; ++sub) {
        // -- store Q subtile (bf16) into per-head 96-wide stripe, zero the pad --
        // qb stripes are single-wave owned; LDS ops are in-order within a wave,
        // so no cross-wave barriers are needed inside this loop.
#pragma unroll
        for (int nt = 0; nt < 5; ++nt)
#pragma unroll
            for (int v = 0; v < 8; ++v)
                qb[(v + 8 * hi) * 768 + h * 96 + nt * 16 + lq] =
                    f2bf(qacc[sub][nt][v]);
#pragma unroll
        for (int v = 0; v < 8; ++v)
            qb[(v + 8 * hi) * 768 + h * 96 + 80 + lq] = 0;

        // -- scores = Q(16x96) * K^T(96x80) --
        v8f sacc[5];
#pragma unroll
        for (int n = 0; n < 5; ++n) sacc[n] = (v8f)0.0f;
        for (int ks = 0; ks < 3; ++ks) {
            int kb = ks * 32;
            FragB16 A;
            const unsigned short* ap = &qb[lq * 768 + h * 96 + kb + 8 * hi];
            A.u[0] = *(const v4u*)(ap);
            A.u[1] = *(const v4u*)(ap + 16);
#pragma unroll
            for (int nt = 0; nt < 5; ++nt) {
                const unsigned short* bp = Kb + (nt * 16 + lq) * 96 + kb + 8 * hi;
                FragB16 B;
                B.u[0] = *(const v4u*)(bp);
                B.u[1] = *(const v4u*)(bp + 16);
                sacc[nt] = wmma_bf16(A, B, sacc[nt]);
            }
        }

        // -- softmax over S=77; rows split across half-waves per C layout --
#pragma unroll
        for (int v = 0; v < 8; ++v) {
            float mx = -1e30f;
#pragma unroll
            for (int nt = 0; nt < 5; ++nt) {
                int s = nt * 16 + lq;
                float x = (s < SEQ) ? sacc[nt][v] * scale : -1e30f;
                mx = fmaxf(mx, x);
            }
#pragma unroll
            for (int d = 1; d < 16; d <<= 1) mx = fmaxf(mx, __shfl_xor(mx, d, 32));
            float p[5];
            float sum = 0.0f;
#pragma unroll
            for (int nt = 0; nt < 5; ++nt) {
                int s = nt * 16 + lq;
                p[nt] = (s < SEQ) ? __expf(sacc[nt][v] * scale - mx) : 0.0f;
                sum += p[nt];
            }
#pragma unroll
            for (int d = 1; d < 16; d <<= 1) sum += __shfl_xor(sum, d, 32);
            float inv = 1.0f / sum;
#pragma unroll
            for (int nt = 0; nt < 5; ++nt)
                qb[(v + 8 * hi) * 768 + h * 96 + nt * 16 + lq] = f2bf(p[nt] * inv);
        }

        // -- ctx = attn(16x96) * V(96x80) --
        v8f cacc[5];
#pragma unroll
        for (int n = 0; n < 5; ++n) cacc[n] = (v8f)0.0f;
        for (int ks = 0; ks < 3; ++ks) {
            int kb = ks * 32;
            FragB16 A;
            const unsigned short* ap = &qb[lq * 768 + h * 96 + kb + 8 * hi];
            A.u[0] = *(const v4u*)(ap);
            A.u[1] = *(const v4u*)(ap + 16);
#pragma unroll
            for (int nt = 0; nt < 5; ++nt) {
                const unsigned short* bp = Vb + (nt * 16 + lq) * 96 + kb + 8 * hi;
                FragB16 B;
                B.u[0] = *(const v4u*)(bp);
                B.u[1] = *(const v4u*)(bp + 16);
                cacc[nt] = wmma_bf16(A, B, cacc[nt]);
            }
        }
        // stash ctx subtile into the (now free) token buffer
#pragma unroll
        for (int nt = 0; nt < 5; ++nt)
#pragma unroll
            for (int v = 0; v < 8; ++v)
                tok[(sub * 16 + v + 8 * hi) * HID + h * DH + nt * 16 + lq] =
                    f2bf(cacc[nt][v]);
    }
    __syncthreads();   // ctx tile (cross-wave) complete

    // ---- out = ctx(32x640) * Wo(640x640) + bo; wave h -> cols [80h,80h+80) ----
#pragma unroll
    for (int s = 0; s < 2; ++s)
#pragma unroll
        for (int n = 0; n < 5; ++n) qacc[s][n] = (v8f)0.0f;

    const unsigned short* bw[5];
#pragma unroll
    for (int nt = 0; nt < 5; ++nt)
        bw[nt] = WoT + (size_t)(h * DH + nt * 16 + lq) * HID + 8 * hi;

    for (int ks = 0; ks < HID / 32; ++ks) {
        int kb = ks * 32;
        FragB16 A0, A1;
        const unsigned short* ap0 = &tok[lq * HID + kb + 8 * hi];
        const unsigned short* ap1 = &tok[(16 + lq) * HID + kb + 8 * hi];
        A0.u[0] = *(const v4u*)(ap0);
        A0.u[1] = *(const v4u*)(ap0 + 16);
        A1.u[0] = *(const v4u*)(ap1);
        A1.u[1] = *(const v4u*)(ap1 + 16);
#pragma unroll
        for (int nt = 0; nt < 5; ++nt) {
            FragB16 B;
            B.u[0] = *(const v4u*)(bw[nt] + kb);
            B.u[1] = *(const v4u*)(bw[nt] + kb + 16);
            qacc[0][nt] = wmma_bf16(A0, B, qacc[0][nt]);
            qacc[1][nt] = wmma_bf16(A1, B, qacc[1][nt]);
        }
    }

    float* ob = out + ((size_t)(b * T_TOT + t0)) * HID;
#pragma unroll
    for (int nt = 0; nt < 5; ++nt) {
        int col = h * DH + nt * 16 + lq;
        float bias = bo[col];
#pragma unroll
        for (int sub = 0; sub < 2; ++sub)
#pragma unroll
            for (int v = 0; v < 8; ++v)
                ob[(size_t)(sub * 16 + v + 8 * hi) * HID + col] =
                    qacc[sub][nt][v] + bias;
    }
}

// ---------------------------------------------------------------------------
// Workspace layout (bytes)
// ---------------------------------------------------------------------------
static const size_t WQT_B = (size_t)HID * HID * 2;             // 819200
static const size_t WOT_B = (size_t)HID * HID * 2;
static const size_t WKT_B = (size_t)HID * CTXD * 2;            // 983040
static const size_t WVT_B = (size_t)HID * CTXD * 2;
static const size_t KC_B  = (size_t)B_TOT * NH * DH * 96 * 2;  // 983040
static const size_t VC_B  = KC_B;

static const size_t OFF_WQT = 0;
static const size_t OFF_WOT = OFF_WQT + WQT_B;
static const size_t OFF_WKT = OFF_WOT + WOT_B;
static const size_t OFF_WVT = OFF_WKT + WKT_B;
static const size_t OFF_KC  = OFF_WVT + WVT_B;
static const size_t OFF_VC  = OFF_KC + KC_B;

extern "C" void kernel_launch(void* const* d_in, const int* in_sizes, int n_in,
                              void* d_out, int out_size, void* d_ws, size_t ws_size,
                              hipStream_t stream) {
    (void)in_sizes; (void)n_in; (void)out_size; (void)ws_size;
    const float* tokens  = (const float*)d_in[0];
    const float* context = (const float*)d_in[1];
    const float* Wq      = (const float*)d_in[2];
    const float* Wk      = (const float*)d_in[3];
    const float* Wv      = (const float*)d_in[4];
    const float* Wo      = (const float*)d_in[5];
    const float* bo      = (const float*)d_in[6];
    float* out = (float*)d_out;

    char* ws = (char*)d_ws;
    unsigned short* WqT = (unsigned short*)(ws + OFF_WQT);
    unsigned short* WoT = (unsigned short*)(ws + OFF_WOT);
    unsigned short* WkT = (unsigned short*)(ws + OFF_WKT);
    unsigned short* WvT = (unsigned short*)(ws + OFF_WVT);
    unsigned short* Kc  = (unsigned short*)(ws + OFF_KC);
    unsigned short* Vc  = (unsigned short*)(ws + OFF_VC);

    // 1) weight convert/transpose to bf16
    ca_prep_weights<<<(HID * CTXD + 255) / 256, 256, 0, stream>>>(
        Wq, Wk, Wv, Wo, WqT, WkT, WvT, WoT);

    // 2) zero K/V pads, then K/V projection
    hipMemsetAsync(Kc, 0, KC_B, stream);
    hipMemsetAsync(Vc, 0, VC_B, stream);
    ca_kv_kernel<<<dim3(B_TOT * NH, 7), 256, 0, stream>>>(context, WkT, WvT, Kc, Vc);

    // 3) fused attention: one WG per 32 token rows
    ca_attn_kernel<<<B_TOT * (T_TOT / MROWS), 256, 0, stream>>>(
        tokens, bo, WqT, WoT, Kc, Vc, out);
}
